// RefineMaskHead_10977936408902
// MI455X (gfx1250) — matrix operand order repaired and checked by reading, not compiled, not run on hardware
//
#include <hip/hip_runtime.h>

// ---------------------------------------------------------------------------
// RefineMask head for MI455X (gfx1250), channel-last (NHWC) f16 pipeline.
// All GEMM-like work uses V_WMMA_F32_16X16X32_F16 with f32 accumulate:
//   A = weights [M x K] (K-contiguous, 3x3 repacked as (tap,ci)),
//   B = activations, channel-last -> contiguous 16-f16 per-lane loads,
//       3x3 border handled by a zero-page base-pointer select (no masking),
//   D = 8 consecutive output channels per lane -> one 16B store.
// Dual accumulators (K step 64) break the WMMA D->C serial chain.
// Irregular ops (top-k, bilinear sample, scatter, upsample) are VALU kernels.
// ---------------------------------------------------------------------------

typedef _Float16 h16;
typedef _Float16 v16h __attribute__((ext_vector_type(16)));
typedef _Float16 v8h  __attribute__((ext_vector_type(8)));
typedef float    v8f  __attribute__((ext_vector_type(8)));

// -------------------------------- f32 -> f16 -------------------------------
__global__ void cvt_f32_f16(const float* __restrict__ src, h16* __restrict__ dst, long n) {
  long i = (long)blockIdx.x * blockDim.x + threadIdx.x;
  if (i < n) dst[i] = (h16)src[i];
}

// ------------------------------ zero fill (f16) ----------------------------
__global__ void zero_f16(h16* __restrict__ dst, long n) {
  long i = (long)blockIdx.x * blockDim.x + threadIdx.x;
  if (i < n) dst[i] = (h16)0.f;
}

// ------------------- NCHW f32 -> NHWC f16 (activations) --------------------
__global__ void nchw2nhwc_f16(const float* __restrict__ src, h16* __restrict__ dst,
                              int Cc, int HW, long total) {
  long i = (long)blockIdx.x * blockDim.x + threadIdx.x;  // dst index
  if (i >= total) return;
  int  c = (int)(i % Cc);
  long r = i / Cc;
  int  p = (int)(r % HW);
  long b = r / HW;
  dst[i] = (h16)src[(b * Cc + c) * HW + p];
}

// ---- conv3x3 weight repack: src[o][ci][ky][kx] -> dst[o][tap*Cin + ci] ----
__global__ void repack_w3(const float* __restrict__ src, h16* __restrict__ dst,
                          int Cin, long total) {
  long i = (long)blockIdx.x * blockDim.x + threadIdx.x;  // dst index
  if (i >= total) return;
  int  ci  = (int)(i % Cin);
  long r   = i / Cin;
  int  tap = (int)(r % 9);
  long o   = r / 9;
  int  ky  = tap / 3, kx = tap - ky * 3;
  dst[i] = (h16)src[((o * Cin + ci) * 3 + ky) * 3 + kx];
}

// ---------------------- WMMA implicit-im2col GEMM kernel -------------------
__device__ __forceinline__ v8f wmma_step(const h16* __restrict__ pa,
                                         const h16* __restrict__ pb, v8f acc) {
  v8h alo = *(const v8h*)(pa);
  v8h ahi = *(const v8h*)(pa + 8);
  v8h blo = *(const v8h*)(pb);
  v8h bhi = *(const v8h*)(pb + 8);
  v16h a, b;
#pragma unroll
  for (int e = 0; e < 8; ++e) {
    a[e] = alo[e]; a[8 + e] = ahi[e];
    b[e] = blo[e]; b[8 + e] = bhi[e];
  }
  return __builtin_amdgcn_wmma_f32_16x16x32_f16(
      false, a, false, b, (short)0, acc, false, false);
}

// Activations channel-last. Out[p][m] = act( sum_k W[m,k]*B[k,p] + bias[m] ).
//   ks==1 : B[k,p] = in[p*K + k]
//   ks==3 : K = (tap, ci); zero-padded SAME conv via zero-page pointer select
// 4 waves/block share the M tile (A reuse), each owns one 16-wide N tile.
// Requires: M%16==0, K%64==0, Cin%64==0 (3x3).
__global__ void __launch_bounds__(128)
wmma_gemm(const h16* __restrict__ Wmat, const h16* __restrict__ inp,
          const float* __restrict__ bias, h16* __restrict__ out16,
          const h16* __restrict__ zeroPage,
          int M, int K, int H, int Wd, int Cin, int ks,
          int Np, long inStrideB, long outStrideB, int ldo, int relu) {
  const int wave  = threadIdx.x >> 5;
  const int lane  = threadIdx.x & 31;
  const int half_ = lane >> 4;   // 0: lanes 0-15, 1: lanes 16-31
  const int r     = lane & 15;
  const int m0    = blockIdx.y * 16;
  const int n0    = (blockIdx.x * 4 + wave) * 16;
  const h16* in   = inp + (long)blockIdx.z * inStrideB;

  int col = n0 + r;
  const bool colOK = col < Np;
  if (!colOK) col = Np - 1;            // clamp: keeps loads valid, EXEC full

  const h16* wrow = Wmat + (long)(m0 + r) * K + half_ * 16;  // A row, this lane
  v8f acc0 = {}, acc1 = {};

  if (ks == 1) {
    const h16* bcol = in + (long)col * K + half_ * 16;
    for (int k0 = 0; k0 < K; k0 += 64) {
      acc0 = wmma_step(wrow + k0,      bcol + k0,      acc0);
      acc1 = wmma_step(wrow + k0 + 32, bcol + k0 + 32, acc1);
    }
  } else {
    const int oy = col / Wd, ox = col - oy * Wd;
    int kbase = 0;
#pragma unroll
    for (int tap = 0; tap < 9; ++tap) {
      const int ky = tap / 3, kx = tap - ky * 3;
      const int iy = oy + ky - 1, ix = ox + kx - 1;
      const bool valid = (unsigned)iy < (unsigned)H && (unsigned)ix < (unsigned)Wd;
      // invalid taps read zeros from the zero page: no per-element masking
      const h16* prow =
          (valid ? (in + ((long)iy * Wd + ix) * Cin) : zeroPage) + half_ * 16;
      for (int ci = 0; ci < Cin; ci += 64) {
        acc0 = wmma_step(wrow + kbase + ci,      prow + ci,      acc0);
        acc1 = wmma_step(wrow + kbase + ci + 32, prow + ci + 32, acc1);
      }
      kbase += Cin;
    }
  }

  if (!colOK) return;
  const int rowb = m0 + half_ * 8;     // D layout: VGPR v -> row rowb+v
  v8h o;
#pragma unroll
  for (int v = 0; v < 8; ++v) {
    float y = acc0[v] + acc1[v] + (bias ? bias[rowb + v] : 0.f);
    if (relu) y = y > 0.f ? y : 0.f;
    o[v] = (h16)y;
  }
  *(v8h*)(out16 + (long)blockIdx.z * outStrideB + (long)col * ldo + rowb) = o;
}

// --------------------------- semantic logits (1ch) -------------------------
__global__ void sem_logits_k(const h16* __restrict__ sem, const float* __restrict__ w,
                             const float* __restrict__ b, float* __restrict__ out,
                             int Cc, long total) {
  long i = (long)blockIdx.x * blockDim.x + threadIdx.x;  // b*HW + p
  if (i >= total) return;
  const h16* s = sem + i * Cc;        // channel-last: contiguous
  float acc = b[0];
  for (int c = 0; c < Cc; ++c) acc += w[c] * (float)s[c];
  out[i] = acc;
}

// ---------------- label-selected 1x1 conv (one class row / roi) ------------
__global__ void label_conv_k(const h16* __restrict__ feats, const float* __restrict__ w,
                             const float* __restrict__ bias, const int* __restrict__ labels,
                             float* __restrict__ out, int cin, int Np, long total) {
  long i = (long)blockIdx.x * blockDim.x + threadIdx.x;  // n*Np + p
  if (i >= total) return;
  int n = (int)(i / Np);
  int l = labels[n];
  const h16* f = feats + i * cin;     // channel-last: contiguous
  const float* ww = w + (long)l * cin;
  float acc = bias[l];
  for (int c = 0; c < cin; ++c) acc += ww[c] * (float)f[c];
  out[i] = acc;
}

// --------------------- top-K via bitonic sort (per roi) --------------------
template <int S>
__global__ void topk_k(const float* __restrict__ pmap, int* __restrict__ idxOut,
                       int Np, int P) {
  __shared__ float vals[S];
  __shared__ int   ids[S];
  int n = blockIdx.x;
  for (int t = threadIdx.x; t < S; t += blockDim.x) {
    vals[t] = (t < Np) ? pmap[(long)n * Np + t] : -3.4e38f;
    ids[t]  = t;
  }
  __syncthreads();
  for (int k = 2; k <= S; k <<= 1) {
    for (int j = k >> 1; j > 0; j >>= 1) {
      for (int t = threadIdx.x; t < S; t += blockDim.x) {
        int ixj = t ^ j;
        if (ixj > t) {
          bool desc = ((t & k) == 0);
          float a = vals[t], b = vals[ixj];
          if (desc ? (a < b) : (a > b)) {
            vals[t] = b; vals[ixj] = a;
            int ti = ids[t]; ids[t] = ids[ixj]; ids[ixj] = ti;
          }
        }
      }
      __syncthreads();
    }
  }
  for (int t = threadIdx.x; t < P; t += blockDim.x)
    idxOut[(long)n * P + t] = ids[t];
}

// ------- bilinear point-sample from sem + coarse gather -> X matrices ------
// semS: (Bimg*40000, cin) channel-last; feats: (N*Np, cin) channel-last.
// X layout: (NPt, 2cin) channel-last: [fine | coarse] per point.
__global__ void gather_points_k(const h16* __restrict__ semS, const h16* __restrict__ feats,
                                const float* __restrict__ rois, const int* __restrict__ idx,
                                h16* __restrict__ Xa, h16* __restrict__ Xb,
                                int cin, int Hm, int Wm, int Np, int NPt) {
  int nj = blockIdx.x;               // n*196 + j
  int n  = nj / 196;
  int ix = idx[nj];
  float px = (0.5f + (float)(ix % Wm)) / (float)Wm;
  float py = (0.5f + (float)(ix / Wm)) / (float)Hm;
  const float* rr = rois + n * 5;
  int   bidx = (int)rr[0];
  float xs = (rr[1] + px * (rr[3] - rr[1])) * 0.25f - 0.5f;
  float ys = (rr[2] + py * (rr[4] - rr[2])) * 0.25f - 0.5f;
  float x0f = floorf(xs), y0f = floorf(ys);
  float dx = xs - x0f, dy = ys - y0f;
  int x0 = (int)x0f, y0 = (int)y0f;
  bool xv0 = (x0 >= 0 && x0 < 200), xv1 = (x0 + 1 >= 0 && x0 + 1 < 200);
  bool yv0 = (y0 >= 0 && y0 < 200), yv1 = (y0 + 1 >= 0 && y0 + 1 < 200);
  int x0c = min(max(x0, 0), 199), x1c = min(max(x0 + 1, 0), 199);
  int y0c = min(max(y0, 0), 199), y1c = min(max(y0 + 1, 0), 199);
  float w00 = (1.f - dx) * (1.f - dy) * (xv0 && yv0 ? 1.f : 0.f);
  float w01 = dx * (1.f - dy)         * (xv1 && yv0 ? 1.f : 0.f);
  float w10 = (1.f - dx) * dy         * (xv0 && yv1 ? 1.f : 0.f);
  float w11 = dx * dy                 * (xv1 && yv1 ? 1.f : 0.f);
  const long base = (long)bidx * 40000;
  const h16* p00 = semS + (base + y0c * 200 + x0c) * cin;
  const h16* p01 = semS + (base + y0c * 200 + x1c) * cin;
  const h16* p10 = semS + (base + y1c * 200 + x0c) * cin;
  const h16* p11 = semS + (base + y1c * 200 + x1c) * cin;
  const h16* cf  = feats + ((long)n * Np + ix) * cin;
  h16* xa = Xa + (long)nj * (2 * cin);
  h16* xb = Xb + (long)nj * (2 * cin);
  for (int c = threadIdx.x; c < cin; c += blockDim.x) {
    float v = w00 * (float)p00[c] + w01 * (float)p01[c]
            + w10 * (float)p10[c] + w11 * (float)p11[c];
    h16 coarse = cf[c];
    xa[c]       = (h16)v;          // fine rows
    xa[cin + c] = coarse;          // coarse rows
    xb[cin + c] = coarse;          // Xb fine rows are fc0's output
  }
}

// ----------------------- scatter refined points back -----------------------
__global__ void scatter_k(const h16* __restrict__ Xf, const int* __restrict__ idx,
                          h16* __restrict__ refined, int cin, int Np, long total) {
  long i = (long)blockIdx.x * blockDim.x + threadIdx.x;   // nj*cin + c
  if (i >= total) return;
  int c  = (int)(i % cin);
  int nj = (int)(i / cin);
  int n  = nj / 196;
  refined[((long)n * Np + idx[nj]) * cin + c] = Xf[i];
}

// ------------------- 2x bilinear upsample + relu (NHWC) --------------------
__global__ void upsample2x_k(const h16* __restrict__ in, h16* __restrict__ out,
                             int Cm, int H, int W, long total) {
  long i = (long)blockIdx.x * blockDim.x + threadIdx.x;   // ((n*Ho+oy)*Wo+ox)*Cm + c
  if (i >= total) return;
  int Ho = 2 * H, Wo = 2 * W;
  int  c  = (int)(i % Cm);
  long r  = i / Cm;
  int  ox = (int)(r % Wo); r /= Wo;
  int  oy = (int)(r % Ho);
  long n  = r / Ho;
  float fy = (oy + 0.5f) * 0.5f - 0.5f;
  float fx = (ox + 0.5f) * 0.5f - 0.5f;
  int y0 = (int)floorf(fy), x0 = (int)floorf(fx);
  float dy = fy - y0, dx = fx - x0;
  int y0c = min(max(y0, 0), H - 1), y1c = min(max(y0 + 1, 0), H - 1);
  int x0c = min(max(x0, 0), W - 1), x1c = min(max(x0 + 1, 0), W - 1);
  const h16* p = in + n * (long)H * W * Cm;
  float v = (1.f - dy) * ((1.f - dx) * (float)p[((long)y0c * W + x0c) * Cm + c]
                        +        dx  * (float)p[((long)y0c * W + x1c) * Cm + c])
          +        dy  * ((1.f - dx) * (float)p[((long)y1c * W + x0c) * Cm + c]
                        +        dx  * (float)p[((long)y1c * W + x1c) * Cm + c]);
  v = v > 0.f ? v : 0.f;
  out[i] = (h16)v;
}

// ===========================================================================
extern "C" void kernel_launch(void* const* d_in, const int* in_sizes, int n_in,
                              void* d_out, int out_size, void* d_ws, size_t ws_size,
                              hipStream_t stream) {
  (void)in_sizes; (void)n_in; (void)out_size; (void)ws_size;
  const int Bimg = 2, N = 128, C = 256, HW = 200 * 200, P = 196;
  const int NPt = N * P;
  const int cins[3]  = {256, 128, 64};
  const int couts[3] = {128, 64, 32};
  const int Hms[3]   = {14, 28, 56};

  float* out = (float*)d_out;
  const size_t OFF_IP[3] = {0, 50176, 250880};
  const size_t OFF_DP[3] = {25088, 150528, 652288};
  const size_t OFF_FI = 1053696, OFF_FD = 2659328, OFF_SEM = 4264960;

  const float* rois   = (const float*)d_in[2];
  const int*   labels = (const int*)d_in[3];

  size_t off = 0;
  auto alloc = [&](size_t bytes) -> void* {
    void* p = (char*)d_ws + off;
    off += (bytes + 255) & ~(size_t)255;
    return p;
  };
  auto cvtNew = [&](const void* src, size_t n) -> h16* {
    h16* d = (h16*)alloc(n * sizeof(h16));
    cvt_f32_f16<<<(int)((n + 255) / 256), 256, 0, stream>>>((const float*)src, d, (long)n);
    return d;
  };
  auto repack3 = [&](const void* src, int Cout, int Cin) -> h16* {
    long n = (long)Cout * 9 * Cin;
    h16* d = (h16*)alloc((size_t)n * sizeof(h16));
    repack_w3<<<(int)((n + 255) / 256), 256, 0, stream>>>((const float*)src, d, Cin, n);
    return d;
  };
  auto toNHWC = [&](const void* src, h16* dst, int Cc, int hw, long total) {
    nchw2nhwc_f16<<<(int)((total + 255) / 256), 256, 0, stream>>>(
        (const float*)src, dst, Cc, hw, total);
  };

  // ---- zero page for conv border reads (>= Cin_max + 64 f16) ----
  h16* zpage = (h16*)alloc(1024 * sizeof(h16));
  zero_f16<<<4, 256, 0, stream>>>(zpage, 1024);

  // ---- weights ----
  h16* instW16[2]; const float* instB[2];
  instW16[0] = repack3(d_in[8],  256, 256); instB[0] = (const float*)d_in[9];
  instW16[1] = repack3(d_in[10], 256, 256); instB[1] = (const float*)d_in[11];
  h16* semW16[4]; const float* semB[4];
  for (int i = 0; i < 4; ++i) {
    semW16[i] = repack3(d_in[12 + 2 * i], 256, 256);
    semB[i]   = (const float*)d_in[13 + 2 * i];
  }
  struct StageP {
    h16 *sem_in_w, *fc_w[3], *fuse_w;
    const float *sem_in_b, *fc_b[3], *fuse_b, *inst_w, *inst_b, *det_w, *det_b;
  } sp[3];
  for (int s = 0; s < 3; ++s) {
    int base = 22 + 14 * s;
    int cin = cins[s], cout = couts[s];
    sp[s].det_b = (const float*)d_in[base + 0];
    sp[s].det_w = (const float*)d_in[base + 1];
    for (int i = 0; i < 3; ++i) sp[s].fc_b[i] = (const float*)d_in[base + 2 + i];
    for (int i = 0; i < 3; ++i) sp[s].fc_w[i] = cvtNew(d_in[base + 5 + i], (size_t)cin * 2 * cin);
    sp[s].fuse_b   = (const float*)d_in[base + 8];
    sp[s].fuse_w   = cvtNew(d_in[base + 9], (size_t)cout * cin);
    sp[s].inst_b   = (const float*)d_in[base + 10];
    sp[s].inst_w   = (const float*)d_in[base + 11];
    sp[s].sem_in_b = (const float*)d_in[base + 12];
    sp[s].sem_in_w = cvtNew(d_in[base + 13], (size_t)cin * 256);
  }

  // ---- activation scratch (all channel-last f16) ----
  const size_t semSz   = (size_t)Bimg * C * HW;
  const size_t featsSz = (size_t)N * 32 * 112 * 112;
  h16* semPing = (h16*)alloc(semSz * sizeof(h16));
  h16* semPong = (h16*)alloc(semSz * sizeof(h16));
  h16* semS    = (h16*)alloc(semSz * sizeof(h16));
  h16* featsA  = (h16*)alloc(featsSz * sizeof(h16));
  h16* featsB  = (h16*)alloc(featsSz * sizeof(h16));
  h16* Xa      = (h16*)alloc((size_t)512 * NPt * sizeof(h16));
  h16* Xb      = (h16*)alloc((size_t)512 * NPt * sizeof(h16));
  h16* Xf      = (h16*)alloc((size_t)256 * NPt * sizeof(h16));
  int* idxBuf  = (int*)alloc((size_t)NPt * sizeof(int));

  auto gemm = [&](const h16* W, const h16* inp, const float* bias, h16* o16,
                  int M, int K, int H, int Wd, int Cin, int ks,
                  int Np, long inS, long outS, int ldo, int relu, int batch) {
    int tilesN = (Np + 15) / 16;
    dim3 grid((tilesN + 3) / 4, M / 16, batch);
    wmma_gemm<<<grid, 128, 0, stream>>>(W, inp, bias, o16, zpage, M, K, H, Wd, Cin, ks,
                                        Np, inS, outS, ldo, relu);
  };

  // ================= semantic tower: 4x conv3x3 + logits =================
  h16 *cur = semPing, *nxt = semPong;
  toNHWC(d_in[1], cur, 256, HW, (long)semSz);
  for (int i = 0; i < 4; ++i) {
    gemm(semW16[i], cur, semB[i], nxt, 256, 2304, 200, 200, 256, 3,
         HW, (long)C * HW, (long)C * HW, 256, 1, Bimg);
    h16* t = cur; cur = nxt; nxt = t;
  }
  h16* semF = cur;
  sem_logits_k<<<(Bimg * HW + 255) / 256, 256, 0, stream>>>(
      semF, (const float*)d_in[21], (const float*)d_in[20], out + OFF_SEM, 256,
      (long)Bimg * HW);

  // ================= instance tower: 2x conv3x3 =================
  h16* ifeat = featsA;
  toNHWC(d_in[0], ifeat, 256, 196, (long)N * C * 196);
  {
    h16* tmp = featsB;
    for (int i = 0; i < 2; ++i) {
      gemm(instW16[i], ifeat, instB[i], tmp, 256, 2304, 14, 14, 256, 3,
           196, (long)C * 196, (long)C * 196, 256, 1, N);
      h16* t = ifeat; ifeat = tmp; tmp = t;
    }
  }

  // ================= 3 refinement stages =================
  for (int s = 0; s < 3; ++s) {
    int cin = cins[s], cout = couts[s], Hm = Hms[s], Np = Hm * Hm;

    // stage semantic features: relu(conv1x1(semF)) -> semS (HW, cin)
    gemm(sp[s].sem_in_w, semF, sp[s].sem_in_b, semS, cin, 256, 200, 200, 256, 1,
         HW, (long)C * HW, (long)cin * HW, cin, 1, Bimg);

    // label-selected instance / detail predictions -> d_out
    long tot = (long)N * Np;
    float* ip = out + OFF_IP[s];
    float* dp = out + OFF_DP[s];
    label_conv_k<<<(int)((tot + 255) / 256), 256, 0, stream>>>(
        ifeat, sp[s].inst_w, sp[s].inst_b, labels, ip, cin, Np, tot);
    label_conv_k<<<(int)((tot + 255) / 256), 256, 0, stream>>>(
        ifeat, sp[s].det_w, sp[s].det_b, labels, dp, cin, Np, tot);

    // top-196 points per roi on instance_preds (bitonic, LDS)
    if (s == 0)      topk_k<256><<<N, 256, 0, stream>>>(ip, idxBuf, Np, P);
    else if (s == 1) topk_k<1024><<<N, 256, 0, stream>>>(ip, idxBuf, Np, P);
    else             topk_k<4096><<<N, 256, 0, stream>>>(ip, idxBuf, Np, P);

    // build X = [fine | coarse] (NPt, 2cin)
    gather_points_k<<<NPt, 128, 0, stream>>>(semS, ifeat, rois, idxBuf,
                                             Xa, Xb, cin, Hm, Hm, Np, NPt);

    // 3x FC with re-concat (write into fine rows of the other X buffer)
    gemm(sp[s].fc_w[0], Xa, sp[s].fc_b[0], Xb, cin, 2 * cin, 1, NPt, 2 * cin, 1,
         NPt, 0, 0, 2 * cin, 1, 1);
    gemm(sp[s].fc_w[1], Xb, sp[s].fc_b[1], Xa, cin, 2 * cin, 1, NPt, 2 * cin, 1,
         NPt, 0, 0, 2 * cin, 1, 1);
    gemm(sp[s].fc_w[2], Xa, sp[s].fc_b[2], Xf, cin, 2 * cin, 1, NPt, 2 * cin, 1,
         NPt, 0, 0, cin, 1, 1);

    // refined = copy(coarse); refined[idx] = Xf
    h16* refined = (ifeat == featsA) ? featsB : featsA;
    hipMemcpyAsync(refined, ifeat, (size_t)N * cin * Np * sizeof(h16),
                   hipMemcpyDeviceToDevice, stream);
    long stot = (long)NPt * cin;
    scatter_k<<<(int)((stot + 255) / 256), 256, 0, stream>>>(Xf, idxBuf, refined, cin, Np, stot);

    // fuse conv1x1 + relu, then 2x bilinear upsample (+relu)
    h16* fused = ifeat;
    gemm(sp[s].fuse_w, refined, sp[s].fuse_b, fused, cout, cin, Hm, Hm, cin, 1,
         Np, (long)cin * Np, (long)cout * Np, cout, 1, N);
    h16* up = refined;
    long utot = (long)N * cout * 4 * Np;
    upsample2x_k<<<(int)((utot + 255) / 256), 256, 0, stream>>>(fused, up, cout, Hm, Hm, utot);
    ifeat = up;
  }

  // ================= final label-selected predictions =================
  {
    long tot = (long)N * 12544;
    label_conv_k<<<(int)((tot + 255) / 256), 256, 0, stream>>>(
        ifeat, (const float*)d_in[7], (const float*)d_in[6], labels, out + OFF_FI, 32, 12544, tot);
    label_conv_k<<<(int)((tot + 255) / 256), 256, 0, stream>>>(
        ifeat, (const float*)d_in[5], (const float*)d_in[4], labels, out + OFF_FD, 32, 12544, tot);
  }
}
